// DreamAttention_47433618817413
// MI455X (gfx1250) — matrix operand order repaired and checked by reading, not compiled, not run on hardware
//
#include <hip/hip_runtime.h>
#include <hip/hip_bf16.h>

// ---------------------------------------------------------------------------
// DreamAttention for MI455X (gfx1250): bf16 WMMA (f32 accumulate) everywhere.
//   B=2 S=1024 HIDDEN=3584 H=28 Hkv=4 D=128 cache=3072 -> T=4096
// Round 5:
//  * __sincosf in fused-RoPE epilogues (hardware v_sin/v_cos, ~10x smaller
//    than precise trig; was the source of the valu/salu static bloat)
//  * flash V staging via b64 loads (d-pair per thread), fewer vmem ops
//  * copy_cache 2D grid (no integer division)
// Kept from round 4: double-buffered LDS pipelines (1 barrier per chunk),
// flash K/V buffer 1 overlaying startup-only Q staging, RoPE+scatter fused
// into projection GEMM epilogues (template modes).
// ---------------------------------------------------------------------------

typedef float        v8f    __attribute__((ext_vector_type(8)));
typedef __bf16       v16bf  __attribute__((ext_vector_type(16)));
typedef __bf16       bf16x2 __attribute__((ext_vector_type(2)));
typedef unsigned int v8u    __attribute__((ext_vector_type(8)));

#define BATCH   2
#define SEQ     1024
#define HID     3584
#define NHEAD   28
#define NKV     4
#define HDIM    128
#define CACHE   3072
#define TTOT    4096
#define MROWS   2048          // B*S
#define LN_THETA 13.815510557964274f   // ln(1e6)

// f32 pair -> packed bf16x2 (RTNE fptrunc -> v_cvt_pk_bf16_f32)
__device__ __forceinline__ unsigned int pack2(float a, float b) {
  bf16x2 v;
  v[0] = (__bf16)a;
  v[1] = (__bf16)b;
  return __builtin_bit_cast(unsigned int, v);
}

// Load a 16-bf16 fragment (8 VGPRs) from an LDS row as two b128 reads.
// `row` must be 16-byte aligned; o0/o1 are u32 offsets (multiples of 4).
__device__ __forceinline__ v16bf load_frag(const unsigned int* row, int o0, int o1) {
  uint4 a = *(const uint4*)(row + o0);
  uint4 b = *(const uint4*)(row + o1);
  v8u r;
  r[0] = a.x; r[1] = a.y; r[2] = a.z; r[3] = a.w;
  r[4] = b.x; r[5] = b.y; r[6] = b.z; r[7] = b.w;
  return __builtin_bit_cast(v16bf, r);
}

__device__ __forceinline__ v8f wmma_bf16(v16bf a, v16bf b, v8f c) {
  return __builtin_amdgcn_wmma_f32_16x16x32_bf16(false, a, false, b, (short)0, c,
                                                 false, false);
}

// ---------------------------------------------------------------------------
// GEMM C = A[M,K] @ W[N,K]^T + bias, 128x128 block tile, 8 waves.
// MODE 0: plain store to C[M,N]
// MODE 1: RoPE(head pairs d, d+64) then store to C[M,N]         (q-proj)
// MODE 2: RoPE then scatter into new_k[b,hk,CACHE+s,d]          (k-proj)
// MODE 3: scatter into new_v[b,hk,CACHE+s,d]                    (v-proj)
// N-tile (128) == HEAD_DIM so sub-tiles j and j+4 hold rotate-half partners.
// Double-buffered LDS: one barrier per 32-K chunk.
// ---------------------------------------------------------------------------
template <int MODE>
__global__ __launch_bounds__(256) void gemm_wmma(
    const float* __restrict__ A, const float* __restrict__ W,
    const float* __restrict__ bias, const int* __restrict__ positions,
    float* __restrict__ C, int M, int N, int K) {
  __shared__ unsigned int lA[2][128][20];
  __shared__ unsigned int lW[2][128][20];

  const int tid  = threadIdx.x;
  const int lane = tid & 31;
  const int wv   = tid >> 5;            // 0..7
  const int m0   = blockIdx.y * 128;
  const int n0   = blockIdx.x * 128;
  const int mq   = lane & 15;           // A-frag row / B-frag col
  const int kh   = lane >> 4;           // K-half select

  v8f acc[8];
#pragma unroll
  for (int j = 0; j < 8; ++j)
#pragma unroll
    for (int r = 0; r < 8; ++r) acc[j][r] = 0.f;

  const int pr = tid & 15;              // pair (2 K-elems) column
  const int rb = tid >> 4;              // row base 0..15

  float2 ra[8], rw[8];
  auto load_chunk = [&](int k0) {
#pragma unroll
    for (int i = 0; i < 8; ++i)
      ra[i] = *(const float2*)(A + (size_t)(m0 + rb + i * 16) * K + k0 + pr * 2);
#pragma unroll
    for (int i = 0; i < 8; ++i)
      rw[i] = *(const float2*)(W + (size_t)(n0 + rb + i * 16) * K + k0 + pr * 2);
  };
  auto commit_chunk = [&](int buf) {
#pragma unroll
    for (int i = 0; i < 8; ++i) lA[buf][rb + i * 16][pr] = pack2(ra[i].x, ra[i].y);
#pragma unroll
    for (int i = 0; i < 8; ++i) lW[buf][rb + i * 16][pr] = pack2(rw[i].x, rw[i].y);
  };

  load_chunk(0);
  commit_chunk(0);
  int cur = 0;
  for (int k0 = 0; k0 < K; k0 += 32) {
    __syncthreads();                    // buf[cur] visible; prev reads done
    const bool nxt = (k0 + 32 < K);
    if (nxt) load_chunk(k0 + 32);       // overlaps the WMMAs below

    // A fragment: lane kh=0 -> K{0..7,16..23}; kh=1 -> K{8..15,24..31}
    v16bf af = load_frag(&lA[cur][wv * 16 + mq][0], kh * 4, 8 + kh * 4);
    v16bf bf[8];
#pragma unroll
    for (int j = 0; j < 8; ++j)
      bf[j] = load_frag(&lW[cur][j * 16 + mq][0], kh * 4, 8 + kh * 4);
#pragma unroll
    for (int j = 0; j < 8; ++j) acc[j] = wmma_bf16(af, bf[j], acc[j]);

    if (nxt) commit_chunk(cur ^ 1);     // write other buffer; no extra barrier
    cur ^= 1;
  }

  const int n    = lane & 15;
  const int hi   = lane >> 4;
  const int row0 = m0 + wv * 16 + 8 * hi;
  float bvs[8];
  if (bias) {
#pragma unroll
    for (int j = 0; j < 8; ++j) bvs[j] = bias[n0 + j * 16 + n];
  } else {
#pragma unroll
    for (int j = 0; j < 8; ++j) bvs[j] = 0.f;
  }

  if (MODE == 0) {
#pragma unroll
    for (int j = 0; j < 8; ++j) {
      float* cp = C + (size_t)row0 * N + (n0 + j * 16 + n);
#pragma unroll
      for (int r = 0; r < 8; ++r) {
        *cp = acc[j][r] + bvs[j];
        cp += N;
      }
    }
  } else {
    float pos[8];
    if (MODE == 1 || MODE == 2) {
#pragma unroll
      for (int r = 0; r < 8; ++r) pos[r] = (float)positions[row0 + r];
    }
#pragma unroll
    for (int j = 0; j < 4; ++j) {       // i = d within head, partner i+64
      const int i = j * 16 + n;         // 0..63  (n0 is a multiple of 128)
      const float freq = __expf(-(float)i * (LN_THETA / 64.f));
#pragma unroll
      for (int r = 0; r < 8; ++r) {
        float x0 = acc[j][r] + bvs[j];
        float x1 = acc[j + 4][r] + bvs[j + 4];
        float o0, o1;
        if (MODE == 3) {
          o0 = x0; o1 = x1;
        } else {
          float sn, c;
          __sincosf(pos[r] * freq, &sn, &c);   // v_sin_f32 / v_cos_f32
          o0 = x0 * c - x1 * sn;
          o1 = x1 * c + x0 * sn;
        }
        const int row = row0 + r;
        if (MODE == 1) {
          float* cp = C + (size_t)row * N + n0 + i;
          cp[0]  = o0;
          cp[64] = o1;
        } else {                        // scatter into new_k / new_v tail
          const int b = row >> 10, s = row & 1023;
          const int hk = n0 >> 7;
          float* cp = C + ((size_t)(b * NKV + hk) * TTOT + CACHE + s) * HDIM + i;
          cp[0]  = o0;
          cp[64] = o1;
        }
      }
    }
  }
}

// ---------------------------------------------------------------------------
// Copy cache into front of new_k / new_v.  blockIdx.y = (b,hk) chunk;
// chunk = 3072*128 f32 contiguous, dst chunk stride 4096*128.
// ---------------------------------------------------------------------------
__global__ __launch_bounds__(256) void copy_cache_kernel(
    float4* __restrict__ dst, const float4* __restrict__ src) {
  const long chunk4 = (long)CACHE * HDIM / 4;           // 98304
  long o = (long)blockIdx.x * 256 + threadIdx.x;        // 0..98303
  long c = blockIdx.y;                                  // 0..7
  dst[c * ((long)TTOT * HDIM / 4) + o] = src[c * chunk4 + o];
}

// ---------------------------------------------------------------------------
// Flash attention.  Block = 256 threads (8 waves); wave owns 16 queries ->
// 128 queries per block.  Double-buffered K/V staging; buffer 1 overlays the
// startup-only Q staging region.  One barrier per 32-key chunk.
// grid.x = (B*H) * (S/128);  per chunk per wave: 8 score WMMA + 8 PV WMMA.
// ---------------------------------------------------------------------------
__global__ __launch_bounds__(256) void flash_attn_kernel(
    const float* __restrict__ q_s, const float* __restrict__ newK,
    const float* __restrict__ newV, float* __restrict__ attn_s) {
  // manual layout (64000 B):
  //   [0      , 10240) Ps  [8][16][20]   per-wave P tile (persistent)
  //   [10240  , 18944) Ks0 [32][68]      K tile buf0
  //   [18944  , 29184) VT0 [128][20]     V^T tile buf0
  //   [29184  , 37888) Ks1 [32][68]      K tile buf1   (overlays Qs)
  //   [37888  , 48128) VT1 [128][20]     V^T tile buf1 (overlays Qs)
  //   [29184  , 64000) Qs  [8][16][68]   startup-only Q staging
  __shared__ __align__(16) unsigned char smem[64000];
  unsigned int* PsB = (unsigned int*)(smem);
  unsigned int* Ks0 = (unsigned int*)(smem + 10240);
  unsigned int* VT0 = (unsigned int*)(smem + 18944);
  unsigned int* Ks1 = (unsigned int*)(smem + 29184);
  unsigned int* VT1 = (unsigned int*)(smem + 37888);
  unsigned int* Qs  = (unsigned int*)(smem + 29184);

  const int tid  = threadIdx.x;
  const int lane = tid & 31;
  const int wv   = tid >> 5;
  const int qb   = blockIdx.x & 7;         // 8 q-blocks of 128
  const int bh   = blockIdx.x >> 3;        // 0..55
  const int b    = bh / NHEAD, h = bh % NHEAD, hkv = h / (NHEAD / NKV);
  const int q0   = qb * 128 + wv * 16;
  const size_t qbase  = ((size_t)(b * SEQ + q0)) * HID + h * HDIM;
  const size_t kvbase = ((size_t)(b * NKV + hkv)) * TTOT * HDIM;

  // ---- stage this wave's 16x128 Q tile (RoPE'd by gemm<1>) as bf16 ----
#pragma unroll
  for (int i = 0; i < 32; ++i) {
    int idx = lane + i * 32;               // 0..1023 pairs
    int row = idx >> 6, p = idx & 63;
    float2 f = *(const float2*)(q_s + qbase + (size_t)row * HID + p * 2);
    Qs[(wv * 16 + row) * 68 + p] = pack2(f.x, f.y);
  }
  const int mq = lane & 15, kh = lane >> 4;
  v16bf qf[4];
#pragma unroll
  for (int c = 0; c < 4; ++c)
    qf[c] = load_frag(&Qs[(wv * 16 + mq) * 68], c * 16 + kh * 4, c * 16 + 8 + kh * 4);

  float mrow[8], lrow[8];
  v8f oacc[8];
#pragma unroll
  for (int r = 0; r < 8; ++r) { mrow[r] = -1e30f; lrow[r] = 0.f; }
#pragma unroll
  for (int j = 0; j < 8; ++j)
#pragma unroll
    for (int r = 0; r < 8; ++r) oacc[j][r] = 0.f;

  const float scale = 0.08838834764831845f;   // 1/sqrt(128)
  const int n = lane & 15, hi = lane >> 4;

  // K staging: thread -> (t = idx>>6, pair p = idx&63), pairs along d.
  // V staging: thread owns a d-pair (dp,dp+1) and 8 consecutive t (b64 loads).
  const int dp = (tid & 63) * 2;
  const int tg = tid >> 6;                 // 0..3 -> t group of 8
  float2 rk[8];
  float2 rv[8];
  auto load_kv = [&](int t0) {
#pragma unroll
    for (int i = 0; i < 8; ++i) {
      int idx = tid + i * 256;
      int t = idx >> 6, p = idx & 63;
      rk[i] = *(const float2*)(newK + kvbase + (size_t)(t0 + t) * HDIM + p * 2);
    }
#pragma unroll
    for (int i = 0; i < 8; ++i)
      rv[i] = *(const float2*)(newV + kvbase + (size_t)(t0 + tg * 8 + i) * HDIM + dp);
  };
  auto commit_kv = [&](unsigned int* ks, unsigned int* vt) {
#pragma unroll
    for (int i = 0; i < 8; ++i) {
      int idx = tid + i * 256;
      int t = idx >> 6, p = idx & 63;
      ks[t * 68 + p] = pack2(rk[i].x, rk[i].y);
    }
#pragma unroll
    for (int ii = 0; ii < 4; ++ii) {       // packed V^T rows dp and dp+1
      vt[dp * 20 + tg * 4 + ii]       = pack2(rv[2 * ii].x, rv[2 * ii + 1].x);
      vt[(dp + 1) * 20 + tg * 4 + ii] = pack2(rv[2 * ii].y, rv[2 * ii + 1].y);
    }
  };

  load_kv(0);
  commit_kv(Ks0, VT0);                     // buf0 does not overlay Qs
  unsigned int *ksA = Ks0, *vtA = VT0, *ksB = Ks1, *vtB = VT1;

  for (int t0 = 0; t0 < TTOT; t0 += 32) {
    __syncthreads();                       // buf A visible; prev reads done
    const bool nxt = (t0 + 32 < TTOT);
    if (nxt) load_kv(t0 + 32);             // overlaps compute below

    // ---- scores: S[16q x 32t] = Q (16x128) . K^T ----
    v8f s0, s1;
#pragma unroll
    for (int r = 0; r < 8; ++r) { s0[r] = 0.f; s1[r] = 0.f; }
    v16bf kb[8];
#pragma unroll
    for (int c = 0; c < 4; ++c) {
      kb[2 * c]     = load_frag(&ksA[n * 68],        c * 16 + kh * 4, c * 16 + 8 + kh * 4);
      kb[2 * c + 1] = load_frag(&ksA[(16 + n) * 68], c * 16 + kh * 4, c * 16 + 8 + kh * 4);
    }
#pragma unroll
    for (int c = 0; c < 4; ++c) {
      s0 = wmma_bf16(qf[c], kb[2 * c],     s0);
      s1 = wmma_bf16(qf[c], kb[2 * c + 1], s1);
    }

    // ---- online softmax (row stats across 16-lane halves) ----
    float corr[8];
#pragma unroll
    for (int r = 0; r < 8; ++r) {
      float a0 = s0[r] * scale, a1 = s1[r] * scale;
      float v = fmaxf(a0, a1);
      v = fmaxf(v, __shfl_xor(v, 1));
      v = fmaxf(v, __shfl_xor(v, 2));
      v = fmaxf(v, __shfl_xor(v, 4));
      v = fmaxf(v, __shfl_xor(v, 8));
      float mn = fmaxf(mrow[r], v);
      float cr = __expf(mrow[r] - mn);
      float p0 = __expf(a0 - mn), p1 = __expf(a1 - mn);
      float ps = p0 + p1;
      ps += __shfl_xor(ps, 1);
      ps += __shfl_xor(ps, 2);
      ps += __shfl_xor(ps, 4);
      ps += __shfl_xor(ps, 8);
      lrow[r] = lrow[r] * cr + ps;
      mrow[r] = mn;
      corr[r] = cr;
      __bf16* prow = (__bf16*)&PsB[(wv * 16 + r + 8 * hi) * 20];
      prow[n]      = (__bf16)p0;
      prow[16 + n] = (__bf16)p1;
    }
#pragma unroll
    for (int j = 0; j < 8; ++j)
#pragma unroll
      for (int r = 0; r < 8; ++r) oacc[j][r] *= corr[r];

    // ---- PV: O[16q x 128d] += P (16x32) . V ----
    v16bf pf = load_frag(&PsB[(wv * 16 + mq) * 20], kh * 4, 8 + kh * 4);
    v16bf vf[8];
#pragma unroll
    for (int j = 0; j < 8; ++j)
      vf[j] = load_frag(&vtA[(j * 16 + n) * 20], kh * 4, 8 + kh * 4);
#pragma unroll
    for (int j = 0; j < 8; ++j) oacc[j] = wmma_bf16(pf, vf[j], oacc[j]);

    if (nxt) commit_kv(ksB, vtB);          // other buffer; no extra barrier
    unsigned int* t;
    t = ksA; ksA = ksB; ksB = t;
    t = vtA; vtA = vtB; vtB = t;
  }

  // ---- epilogue: divide by row sum, write [B,S,H*D] ----
#pragma unroll
  for (int r = 0; r < 8; ++r) lrow[r] = 1.f / lrow[r];
#pragma unroll
  for (int j = 0; j < 8; ++j) {
    float* op = attn_s + ((size_t)(b * SEQ + q0 + 8 * hi)) * HID + h * HDIM + j * 16 + n;
#pragma unroll
    for (int r = 0; r < 8; ++r) {
      *op = oacc[j][r] * lrow[r];
      op += HID;
    }
  }
}

// ---------------------------------------------------------------------------
extern "C" void kernel_launch(void* const* d_in, const int* in_sizes, int n_in,
                              void* d_out, int out_size, void* d_ws, size_t ws_size,
                              hipStream_t stream) {
  const int*   positions = (const int*)d_in[0];
  const float* hidden    = (const float*)d_in[1];
  const float* k_cache   = (const float*)d_in[2];
  const float* v_cache   = (const float*)d_in[3];
  // d_in[4] = cache_len (static 3072)
  const float* q_w = (const float*)d_in[5];
  const float* q_b = (const float*)d_in[6];
  const float* k_w = (const float*)d_in[7];
  const float* k_b = (const float*)d_in[8];
  const float* v_w = (const float*)d_in[9];
  const float* v_b = (const float*)d_in[10];
  const float* o_w = (const float*)d_in[11];

  float* out   = (float*)d_out;
  float* new_k = out + (size_t)BATCH * SEQ * HID;
  float* new_v = new_k + (size_t)BATCH * NKV * TTOT * HDIM;

  float* q_sc  = (float*)d_ws;                        // 2048*3584 f32
  float* at_sc = q_sc + (size_t)MROWS * HID;          // 2048*3584 f32 (~59MB)

  // Projections (bf16 WMMA, f32 accumulate); RoPE/scatter fused in epilogues.
  gemm_wmma<1><<<dim3(HID / 128, MROWS / 128), 256, 0, stream>>>(
      hidden, q_w, q_b, positions, q_sc, MROWS, HID, HID);
  gemm_wmma<2><<<dim3((NKV * HDIM) / 128, MROWS / 128), 256, 0, stream>>>(
      hidden, k_w, k_b, positions, new_k, MROWS, NKV * HDIM, HID);
  gemm_wmma<3><<<dim3((NKV * HDIM) / 128, MROWS / 128), 256, 0, stream>>>(
      hidden, v_w, v_b, nullptr, new_v, MROWS, NKV * HDIM, HID);

  // KV-cache concat front halves
  copy_cache_kernel<<<dim3(CACHE * HDIM / 4 / 256, 8), 256, 0, stream>>>(
      (float4*)new_k, (const float4*)k_cache);
  copy_cache_kernel<<<dim3(CACHE * HDIM / 4 / 256, 8), 256, 0, stream>>>(
      (float4*)new_v, (const float4*)v_cache);

  // Flash attention (8 waves / 128 queries per block)
  flash_attn_kernel<<<dim3((BATCH * NHEAD) * (SEQ / 128)), 256, 0, stream>>>(
      q_sc, new_k, new_v, at_sc);

  // Output projection
  gemm_wmma<0><<<dim3(HID / 128, MROWS / 128), 256, 0, stream>>>(
      at_sc, o_w, nullptr, nullptr, out, MROWS, HID, HID);
}
